// CrossAttention_58643483459602
// MI455X (gfx1250) — matrix operand order repaired and checked by reading, not compiled, not run on hardware
//
#include <hip/hip_runtime.h>
#include <hip/hip_bf16.h>

// ---------------------------------------------------------------------------
// CrossAttention on MI455X (gfx1250): f16 WMMA projections + flash attention.
// B=4, C=256, H=W=64 -> N=4096. GEMMs via v_wmma_f32_16x16x32_f16; K/V tile
// staging via the Tensor Data Mover (tensor_load_to_lds + s_wait_tensorcnt),
// falling back to global_load_async_to_lds_b128, then plain copies.
// ---------------------------------------------------------------------------

typedef __attribute__((ext_vector_type(16))) _Float16 v16h;
typedef __attribute__((ext_vector_type(8)))  _Float16 v8h;
typedef __attribute__((ext_vector_type(8)))  float    v8f;
typedef __attribute__((ext_vector_type(4)))  unsigned int u32x4;
typedef __attribute__((ext_vector_type(8)))  int      i32x8;
typedef __attribute__((ext_vector_type(4)))  int      i32x4;

#define CCH   256      // channels
#define NPIX  4096     // H*W
#define NB    4        // batch

#if __has_builtin(__builtin_amdgcn_tensor_load_to_lds)
#define USE_TDM 1
#elif __has_builtin(__builtin_amdgcn_global_load_async_to_lds_b128)
#define USE_ASYNC 1
#endif

// Build a 16-half A/B fragment from two aligned 8-half chunks.
__device__ __forceinline__ v16h load_frag2(const _Float16* p0, const _Float16* p1) {
    const v8h a = *(const v8h*)p0;
    const v8h b = *(const v8h*)p1;
    v16h r;
#pragma unroll
    for (int i = 0; i < 8; ++i) { r[i] = a[i]; r[i + 8] = b[i]; }
    return r;
}

#ifdef USE_TDM
// Issue one 2D TDM tile load: global (row-major, 2-byte elems, row stride
// `stride_e` elements) -> LDS at byte offset `lds_addr`, with LDS row padding
// (pad codes per ISA Tbl: interval 2^(c+1) dwords, amount (c+1) dwords).
__device__ __forceinline__ void tdm_load_2d(unsigned lds_addr, const void* gsrc,
                                            unsigned tile_w, unsigned tile_h,
                                            unsigned stride_e,
                                            unsigned pad_int_code,
                                            unsigned pad_amt_code)
{
    unsigned long long ga = (unsigned long long)(uintptr_t)gsrc;
    u32x4 g0;
    g0[0] = 1u;                                         // count=1, user mode
    g0[1] = lds_addr;                                   // LDS byte address
    g0[2] = (unsigned)(ga & 0xffffffffull);             // global addr lo
    g0[3] = (unsigned)((ga >> 32) & 0x1ffffffull)       // global addr hi
          | (2u << 30);                                 // type = 2 (image)
    i32x8 g1;
    g1[0] = (int)((1u << 16)                            // data_size = 2 bytes
                | (1u << 20)                            // pad_enable
                | (pad_int_code << 22)
                | (pad_amt_code << 25));
    g1[1] = (int)((tile_w & 0xffffu) << 16);            // tensor_dim0 lo16
    g1[2] = (int)(((tile_w >> 16) & 0xffffu)            // tensor_dim0 hi16
                | ((tile_h & 0xffffu) << 16));          // tensor_dim1 lo16
    g1[3] = (int)(((tile_h >> 16) & 0xffffu)            // tensor_dim1 hi16
                | ((tile_w & 0xffffu) << 16));          // tile_dim0
    g1[4] = (int)(tile_h & 0xffffu);                    // tile_dim1 (dim2 = 0)
    g1[5] = (int)stride_e;                              // dim0_stride lo32
    g1[6] = 0;                                          // stride hi / dim1_stride
    g1[7] = 0;
    i32x4 z4 = {0, 0, 0, 0};
#if __clang_major__ >= 23
    i32x8 z8 = {0, 0, 0, 0, 0, 0, 0, 0};
    __builtin_amdgcn_tensor_load_to_lds(g0, g1, z4, z4, z8, 0);
#else
    __builtin_amdgcn_tensor_load_to_lds(g0, g1, z4, z4, 0);
#endif
}
#endif

#ifdef USE_ASYNC
__device__ __forceinline__ void async_cp16(_Float16* dst, const _Float16* src) {
    __builtin_amdgcn_global_load_async_to_lds_b128(
        (__attribute__((address_space(1))) void*)(uintptr_t)src,
        (__attribute__((address_space(3))) void*)(unsigned)(uintptr_t)dst,
        0, 0);
}
#endif

// ---------------------------------------------------------------------------
// Kernel 0: convert f32 weight matrices to f16 in workspace.
// ---------------------------------------------------------------------------
__global__ void cvt_w_kernel(const float* __restrict__ Wq,
                             const float* __restrict__ Wk,
                             const float* __restrict__ Wv,
                             _Float16* __restrict__ Wqh,
                             _Float16* __restrict__ Wkh,
                             _Float16* __restrict__ Wvh) {
    int i = blockIdx.x * 256 + threadIdx.x;   // CCH*CCH = 65536 elements
    Wqh[i] = (_Float16)Wq[i];
    Wkh[i] = (_Float16)Wk[i];
    Wvh[i] = (_Float16)Wv[i];
}

// ---------------------------------------------------------------------------
// Kernel 1: projections.
//   Qh[b][n][c] = sum_c' Wq[c][c'] x[b][c'][n] + bq[c]     (f16)
//   Kh[b][n][c] = same with Wk,bk,y
//   Vt[b][c][n] = sum_c' Wv[c][c'] y[b][c'][n] + bv[c]     (f16, transposed)
// ---------------------------------------------------------------------------
__global__ __launch_bounds__(256) void proj_kernel(
    const float* __restrict__ x, const float* __restrict__ y,
    const _Float16* __restrict__ Wqh, const float* __restrict__ bq,
    const _Float16* __restrict__ Wkh, const float* __restrict__ bk,
    const _Float16* __restrict__ Wvh, const float* __restrict__ bv,
    _Float16* __restrict__ Qh, _Float16* __restrict__ Kh,
    _Float16* __restrict__ Vt)
{
    const int b    = blockIdx.x >> 5;       // 32 pixel-blocks per batch
    const int p0   = (blockIdx.x & 31) * 128;
    const int tid  = threadIdx.x;
    const int lane = tid & 31;
    const int wave = tid >> 5;              // 0..7 -> 16-pixel strip
    const int lo16 = lane & 15;
    const int hi   = lane >> 4;             // 0 or 1

    // [pixel][k] slice, row stride 40 halves (80B, 16B aligned rows)
    __shared__ _Float16 xs[128 * 40];

    const v8f vzero = {0.f,0.f,0.f,0.f,0.f,0.f,0.f,0.f};

    for (int proj = 0; proj < 3; ++proj) {
        const float*    src  = (proj == 0) ? x   : y;
        const _Float16* Wh   = (proj == 0) ? Wqh : (proj == 1) ? Wkh : Wvh;
        const float*    bias = (proj == 0) ? bq  : (proj == 1) ? bk  : bv;

        v8f acc[16];
#pragma unroll
        for (int t = 0; t < 16; ++t) acc[t] = vzero;

        for (int k = 0; k < 8; ++k) {
            __syncthreads();
            // Stage src[b][k*32+kk][p0+p], kk=0..31, p=0..127 (coalesced).
#pragma unroll
            for (int it = 0; it < 16; ++it) {
                int idx = tid + it * 256;
                int kk  = idx >> 7;
                int p   = idx & 127;
                float v = src[((size_t)b * CCH + (k * 32 + kk)) * NPIX + p0 + p];
                xs[p * 40 + kk] = (_Float16)v;
            }
            __syncthreads();

            // A fragment: pixel row M = lane%16 within this wave's strip.
            const _Float16* arow = &xs[(wave * 16 + lo16) * 40];
            const int ksel = hi * 16;
            v16h a = load_frag2(arow + ksel, arow + ksel + 8);

#pragma unroll
            for (int t = 0; t < 16; ++t) {
                // B fragment: 16 consecutive c from W[o][c] row (32B aligned).
                const _Float16* bp = &Wh[(size_t)(t * 16 + lo16) * CCH + k * 32 + hi * 16];
                v16h bf = *(const v16h*)bp;
                acc[t] = __builtin_amdgcn_wmma_f32_16x16x32_f16(
                    false, a, false, bf, (short)0, acc[t], false, false);
            }
        }

        // Store with bias; D layout: (m = r + 8*hi, n = lane%16).
        const int prow = p0 + wave * 16;
#pragma unroll
        for (int t = 0; t < 16; ++t) {
            int   o  = t * 16 + lo16;
            float bs = bias[o];
#pragma unroll
            for (int r = 0; r < 8; ++r) {
                int   pixel = prow + r + 8 * hi;
                _Float16 hv = (_Float16)(acc[t][r] + bs);
                if (proj == 0)      Qh[((size_t)b * NPIX + pixel) * CCH + o] = hv;
                else if (proj == 1) Kh[((size_t)b * NPIX + pixel) * CCH + o] = hv;
                else                Vt[((size_t)b * CCH + o) * NPIX + pixel] = hv;
            }
        }
    }
}

// ---------------------------------------------------------------------------
// Kernel 2: flash attention (no 1/sqrt(d) scaling, matching reference).
// 4 waves / block, 64 query rows; each wave: 16-row strip with resident Q
// fragments + 16x256 f32 accumulators. 32-key blocks staged into LDS via TDM.
// ---------------------------------------------------------------------------
__global__ __launch_bounds__(128) void flash_kernel(
    const _Float16* __restrict__ Qh, const _Float16* __restrict__ Kh,
    const _Float16* __restrict__ Vt, float* __restrict__ out)
{
    const int b    = blockIdx.x >> 6;      // 64 q-blocks per batch
    const int qb   = blockIdx.x & 63;
    const int tid  = threadIdx.x;
    const int lane = tid & 31;
    const int wave = tid >> 5;             // 0..3
    const int lo16 = lane & 15;
    const int hi   = lane >> 4;

    __shared__ _Float16 Ks [32 * 264];     // [key][c]   row stride 264 halves (528B)
    __shared__ _Float16 Vts[256 * 40];     // [c][key]   row stride 40 halves (80B)
    __shared__ _Float16 Pst[4][16 * 40];   // per-wave P tile [qrow][key]

    // Resident Q fragments: A row M = lane%16 => query row.
    const int qrow = qb * 64 + wave * 16 + lo16;
    const _Float16* qbase = &Qh[((size_t)b * NPIX + qrow) * CCH];
    v16h qf[8];
#pragma unroll
    for (int kk = 0; kk < 8; ++kk) {
        int c0 = kk * 32 + hi * 16;
        qf[kk] = load_frag2(qbase + c0, qbase + c0 + 8);
    }

    const v8f vzero = {0.f,0.f,0.f,0.f,0.f,0.f,0.f,0.f};
    v8f o_acc[16];
#pragma unroll
    for (int t = 0; t < 16; ++t) o_acc[t] = vzero;
    float mrow[8], lrow[8];
#pragma unroll
    for (int r = 0; r < 8; ++r) { mrow[r] = -3.0e38f; lrow[r] = 0.0f; }

    for (int j = 0; j < NPIX / 32; ++j) {
        __syncthreads();
        const _Float16* kg = &Kh[((size_t)b * NPIX + j * 32) * CCH];
        const _Float16* vg = &Vt[(size_t)b * CCH * NPIX + j * 32];

#if defined(USE_TDM)
        if (wave == 0) {
            // K block: 32 rows x 256 elems, global stride 256; LDS pad ->
            //   row stride 528B (interval 2^7=128 dwords, pad 4 dwords).
            tdm_load_2d((unsigned)(uintptr_t)&Ks[0], kg, 256u, 32u, 256u, 6u, 3u);
            // V^T block: 256 rows x 32 elems, global stride 4096; LDS pad ->
            //   row stride 80B (interval 2^4=16 dwords, pad 4 dwords).
            tdm_load_2d((unsigned)(uintptr_t)&Vts[0], vg, 32u, 256u, 4096u, 3u, 3u);
        }
        __builtin_amdgcn_s_wait_tensorcnt(0);
#else
        // Stage K rows 0..15: 16 rows x 256 halves = 512 x 16B chunks.
#pragma unroll
        for (int it = 0; it < 4; ++it) {
            int ci = tid + it * 128;
            int row = ci >> 5, cc = (ci & 31) * 8;
#ifdef USE_ASYNC
            async_cp16(&Ks[row * 264 + cc], &kg[row * CCH + cc]);
#else
            *(v8h*)&Ks[row * 264 + cc] = *(const v8h*)&kg[row * CCH + cc];
#endif
        }
        // Stage V^T block: 256 rows x 32 halves = 1024 x 16B chunks.
#pragma unroll
        for (int it = 0; it < 8; ++it) {
            int ci = tid + it * 128;
            int row = ci >> 2, cc = (ci & 3) * 8;
#ifdef USE_ASYNC
            async_cp16(&Vts[row * 40 + cc], &vg[(size_t)row * NPIX + cc]);
#else
            *(v8h*)&Vts[row * 40 + cc] = *(const v8h*)&vg[(size_t)row * NPIX + cc];
#endif
        }
#if defined(USE_ASYNC)
#if __has_builtin(__builtin_amdgcn_s_wait_asynccnt)
        __builtin_amdgcn_s_wait_asynccnt(0);
#else
        asm volatile("s_wait_asynccnt 0x0" ::: "memory");
#endif
#endif
#endif // USE_TDM

        // Prefetch next key block into L2 (lowers to global_prefetch_b8).
        if (j + 1 < NPIX / 32) {
            const _Float16* kn = &Kh[((size_t)b * NPIX + (j + 1) * 32) * CCH];
            __builtin_prefetch(kn + tid * 64, 0, 1);
        }
        __syncthreads();

        // S = Q * K^T : two 16x16 col tiles.
        v8f s[2];
#pragma unroll
        for (int t = 0; t < 2; ++t) {
            v8f a = vzero;
#pragma unroll
            for (int kk = 0; kk < 8; ++kk) {
                const _Float16* bp = &Ks[(t * 16 + lo16) * 264 + kk * 32 + hi * 16];
                v16h bf = load_frag2(bp, bp + 8);
                a = __builtin_amdgcn_wmma_f32_16x16x32_f16(
                    false, qf[kk], false, bf, (short)0, a, false, false);
            }
            s[t] = a;
        }

        // Online softmax. Row m = r + 8*hi lives across one 16-lane half.
        float mnew[8], scale[8];
#pragma unroll
        for (int r = 0; r < 8; ++r) {
            float mx = fmaxf(s[0][r], s[1][r]);
#pragma unroll
            for (int off = 1; off < 16; off <<= 1)
                mx = fmaxf(mx, __shfl_xor(mx, off, 32));
            mnew[r]  = fmaxf(mrow[r], mx);
            scale[r] = __expf(mrow[r] - mnew[r]);
        }
#pragma unroll
        for (int r = 0; r < 8; ++r) {
            float sum = 0.f;
#pragma unroll
            for (int t = 0; t < 2; ++t) {
                float p = __expf(s[t][r] - mnew[r]);
                s[t][r] = p;
                sum += p;
            }
#pragma unroll
            for (int off = 1; off < 16; off <<= 1)
                sum += __shfl_xor(sum, off, 32);
            lrow[r] = lrow[r] * scale[r] + sum;
            mrow[r] = mnew[r];
        }
#pragma unroll
        for (int t = 0; t < 16; ++t)
#pragma unroll
            for (int r = 0; r < 8; ++r)
                o_acc[t][r] *= scale[r];

        // D-layout -> row-major LDS, then reload as A fragment (same wave).
        _Float16* pw = &Pst[wave][0];
#pragma unroll
        for (int t = 0; t < 2; ++t)
#pragma unroll
            for (int r = 0; r < 8; ++r)
                pw[(r + 8 * hi) * 40 + t * 16 + lo16] = (_Float16)s[t][r];
        asm volatile("s_wait_dscnt 0x0" ::: "memory");

        const _Float16* pr = &pw[lo16 * 40 + hi * 16];
        v16h pf = load_frag2(pr, pr + 8);

        // O += P * V : 16 output col tiles.
#pragma unroll
        for (int t = 0; t < 16; ++t) {
            const _Float16* bp = &Vts[(t * 16 + lo16) * 40 + hi * 16];
            v16h bf = load_frag2(bp, bp + 8);
            o_acc[t] = __builtin_amdgcn_wmma_f32_16x16x32_f16(
                false, pf, false, bf, (short)0, o_acc[t], false, false);
        }
    }

    // Normalize and store: out[b][c][pixel], pixel = base + r + 8*hi.
    float rinv[8];
#pragma unroll
    for (int r = 0; r < 8; ++r) rinv[r] = 1.0f / lrow[r];
    const int prow = qb * 64 + wave * 16;
#pragma unroll
    for (int t = 0; t < 16; ++t) {
        int c = t * 16 + lo16;
        float* ob = &out[((size_t)b * CCH + c) * NPIX + prow];
#pragma unroll
        for (int r = 0; r < 8; ++r)
            ob[r + 8 * hi] = o_acc[t][r] * rinv[r];
    }
}

// ---------------------------------------------------------------------------
extern "C" void kernel_launch(void* const* d_in, const int* in_sizes, int n_in,
                              void* d_out, int out_size, void* d_ws, size_t ws_size,
                              hipStream_t stream) {
    const float* x  = (const float*)d_in[0];
    const float* y  = (const float*)d_in[1];
    const float* Wq = (const float*)d_in[2];
    const float* bq = (const float*)d_in[3];
    const float* Wk = (const float*)d_in[4];
    const float* bk = (const float*)d_in[5];
    const float* Wv = (const float*)d_in[6];
    const float* bv = (const float*)d_in[7];
    float* out = (float*)d_out;

    char* ws = (char*)d_ws;
    const size_t szQKV = (size_t)NB * NPIX * CCH * sizeof(_Float16); // 8 MB each
    _Float16* Qh  = (_Float16*)(ws);
    _Float16* Kh  = (_Float16*)(ws + szQKV);
    _Float16* Vt  = (_Float16*)(ws + 2 * szQKV);
    _Float16* Wqh = (_Float16*)(ws + 3 * szQKV);
    _Float16* Wkh = Wqh + CCH * CCH;
    _Float16* Wvh = Wkh + CCH * CCH;

    cvt_w_kernel<<<(CCH * CCH) / 256, 256, 0, stream>>>(Wq, Wk, Wv, Wqh, Wkh, Wvh);
    proj_kernel<<<NB * 32, 256, 0, stream>>>(x, y, Wqh, bq, Wkh, bk, Wvh, bv,
                                             Qh, Kh, Vt);
    flash_kernel<<<NB * 64, 128, 0, stream>>>(Qh, Kh, Vt, out);
}
